// VideoGlobalResSNDetDiscriminator_962072674493
// MI455X (gfx1250) — matrix operand order repaired and checked by reading, not compiled, not run on hardware
//
#include <hip/hip_runtime.h>
#include <hip/hip_bf16.h>

typedef __bf16 bf16;
typedef __attribute__((ext_vector_type(16))) __bf16 v16bf;
typedef __attribute__((ext_vector_type(8)))  __bf16 v8bf;
typedef __attribute__((ext_vector_type(8)))  float  v8f;
typedef __attribute__((ext_vector_type(4)))  unsigned int v4u;
typedef __attribute__((ext_vector_type(8)))  int v8i;
typedef __attribute__((ext_vector_type(4)))  int v4i;

#define T_ 4

#if defined(__has_builtin)
#if __has_builtin(__builtin_amdgcn_tensor_load_to_lds) && __has_builtin(__builtin_amdgcn_s_wait_tensorcnt)
#define HAVE_TDM 1
#endif
#endif
#ifndef HAVE_TDM
#define HAVE_TDM 0
#endif

// ---------------------------------------------------------------------------
// Patch builder: (16,3,128,128)+(16,6,128,128)+bboxes -> (64,128,128,32) bf16
// NHWC with channels padded 9 -> 32 so K = 9*32 = 288 is a multiple of 32.
// ---------------------------------------------------------------------------
__global__ __launch_bounds__(256) void build_patches(
    const float* __restrict__ x, const float* __restrict__ input_x,
    const int* __restrict__ bboxes, bf16* __restrict__ out)
{
  int idx = blockIdx.x * blockDim.x + threadIdx.x;   // 64*128*128 threads
  int p   = idx >> 14;           // patch 0..63
  int pix = idx & 16383;
  int h = pix >> 7, w = pix & 127;
  int b = p >> 2, t = p & 3;

  bool m[3];
#pragma unroll
  for (int i = 0; i < 3; i++) {
    const int* bb = &bboxes[((b * 3 + i) * T_ + t) * 5];
    m[i] = (h >= bb[1]) && (h <= bb[3]) && (w >= bb[2]) && (w <= bb[4]);
  }

  float vals[32];
#pragma unroll
  for (int c = 0; c < 32; c++) vals[c] = 0.0f;

  size_t sp = (size_t)h * 128 + w;
  if (m[0]) {
#pragma unroll
    for (int c = 0; c < 3; c++) vals[c] = input_x[((size_t)(b * 6 + c) * 16384) + sp];
  }
  if (m[1]) {
#pragma unroll
    for (int c = 0; c < 3; c++) vals[3 + c] = x[((size_t)(b * 3 + c) * 16384) + sp];
  }
  if (m[2]) {
#pragma unroll
    for (int c = 0; c < 3; c++) vals[6 + c] = input_x[((size_t)(b * 6 + 3 + c) * 16384) + sp];
  }

  bf16 o[32];
#pragma unroll
  for (int c = 0; c < 32; c++) o[c] = (bf16)vals[c];

  bf16* dst = &out[(size_t)idx * 32];
#pragma unroll
  for (int k = 0; k < 4; k++) {
    v8bf vv;
#pragma unroll
    for (int j = 0; j < 8; j++) vv[j] = o[k * 8 + j];
    *(v8bf*)&dst[k * 8] = vv;
  }
}

// ---------------------------------------------------------------------------
// Spectral norm + transpose + bf16 quantize.
// sigma = || W @ normalize(W^T u) ||.  Output layout: [cout][ky][kx][cinp].
// ---------------------------------------------------------------------------
__device__ float block_reduce256(float v, float* red) {
  int tid = threadIdx.x;
  red[tid] = v;
  __syncthreads();
  for (int s = 128; s > 0; s >>= 1) {
    if (tid < s) red[tid] += red[tid + s];
    __syncthreads();
  }
  float r = red[0];
  __syncthreads();
  return r;
}

__global__ __launch_bounds__(256) void sn_transpose(
    const float* __restrict__ w, const float* __restrict__ u,
    bf16* __restrict__ wbf, int cout, int cin, int cinp)
{
  __shared__ float vsh[2304];   // max Kc = 256*9
  __shared__ float red[256];
  int tid = threadIdx.x;
  int Kc  = cin * 9;

  float ss = 0.0f;
  for (int j = tid; j < Kc; j += 256) {
    float acc = 0.0f;
    for (int i = 0; i < cout; i++) acc += w[(size_t)i * Kc + j] * u[i];
    vsh[j] = acc;
    ss += acc * acc;
  }
  float vn = sqrtf(block_reduce256(ss, red));
  float inv_vn = 1.0f / (vn + 1e-12f);
  for (int j = tid; j < Kc; j += 256) vsh[j] *= inv_vn;
  __syncthreads();

  float ss2 = 0.0f;
  for (int i = tid; i < cout; i += 256) {
    float acc = 0.0f;
    for (int j = 0; j < Kc; j++) acc += w[(size_t)i * Kc + j] * vsh[j];
    ss2 += acc * acc;
  }
  float yn2 = block_reduce256(ss2, red);
  float yn  = sqrtf(yn2);
  float sigma = yn2 / (yn + 1e-12f);
  float inv = 1.0f / sigma;

  int E = cout * 9 * cinp;
  for (int e = tid; e < E; e += 256) {
    int o   = e / (9 * cinp);
    int rem = e - o * 9 * cinp;
    int kyx = rem / cinp;
    int c   = rem - kyx * cinp;
    float val = (c < cin) ? w[((size_t)o * cin + c) * 9 + kyx] * inv : 0.0f;
    wbf[e] = (bf16)val;
  }
}

// ---------------------------------------------------------------------------
// Implicit-GEMM 3x3 conv (pad=1), NHWC bf16, f32 accumulate via WMMA.
// Tile M=128 x N=64, 8 waves, 32x32 per wave => 4 WMMAs per 32-wide K-chunk.
// Double-buffered LDS, one barrier per chunk, incremental (ky,kx,c) decode.
// B (weight) tile is DMA'd by the Tensor Data Mover: 64x32 bf16 tile, LDS
// row pitch 80B produced via D# padding (interval=16 DW, amount=4 DW).
// ---------------------------------------------------------------------------
#define BM 128
#define BN 64
#define BK 32
#define LDK 40   // row pitch in bf16 (80B = 20-bank stride, conflict-free)

#if HAVE_TDM
__device__ __forceinline__ void tdm_load_b_tile(const bf16* gptr, void* ldsptr,
                                                int K, int COUT) {
  unsigned long long ga = (unsigned long long)(__SIZE_TYPE__)gptr;
  v4u g0;
  g0[0] = 1u;                                   // count=1, user descriptor
  g0[1] = (unsigned int)(__SIZE_TYPE__)ldsptr;  // lds_addr (low 32 bits of flat)
  g0[2] = (unsigned int)ga;                     // global_addr[31:0]
  g0[3] = (unsigned int)(ga >> 32) | (2u << 30);// global_addr[56:32] | type=2
  v8i g1;
  // data_size=1(2B), pad_enable, pad_interval=3(16 DW), pad_amount=3(4 DW)
  g1[0] = (int)((1u << 16) | (1u << 20) | (3u << 22) | (3u << 25));
  g1[1] = (int)(((unsigned)K & 0xFFFFu) << 16);              // tensor_dim0 lo
  g1[2] = (int)(((unsigned)K >> 16) | (((unsigned)COUT & 0xFFFFu) << 16)); // dim0 hi | dim1 lo
  g1[3] = (int)(32u << 16);                                  // dim1 hi=0 | tile_dim0=32
  g1[4] = 64;                                                // tile_dim1=64, tile_dim2=0
  g1[5] = K;                                                 // dim0_stride[31:0]
  g1[6] = 0;                                                 // dim0_stride hi | dim1_stride lo
  g1[7] = 0;
  v4i zz;
  zz[0] = 0; zz[1] = 0; zz[2] = 0; zz[3] = 0;                // <=2D: groups 2/3 zero
  v8i zz8;
  zz8[0] = 0; zz8[1] = 0; zz8[2] = 0; zz8[3] = 0;
  zz8[4] = 0; zz8[5] = 0; zz8[6] = 0; zz8[7] = 0;
  // 6-arg form (clang-23 / therock headers): extra int32x8 group before cpol
  __builtin_amdgcn_tensor_load_to_lds(g0, g1, zz, zz, zz8, 0);
}
#endif

template <bool LRELU, bool RESID>
__global__ __launch_bounds__(256) void conv3x3_wmma(
    const bf16* __restrict__ in, const bf16* __restrict__ wgt,
    const float* __restrict__ bias, const bf16* __restrict__ resid,
    bf16* __restrict__ out,
    int IH, int IW, int OH, int OW, int CIN, int COUT, int stride)
{
  __shared__ __attribute__((aligned(16))) bf16 As[2][BM * LDK];
  __shared__ __attribute__((aligned(16))) bf16 Bs[2][BN * LDK];

  const int K    = CIN * 9;
  const int tid  = threadIdx.x;
  const int lane = tid & 31;
  const int wv   = tid >> 5;
  const int mw   = (wv & 3) * 32;   // wave M offset in tile
  const int nw   = (wv >> 2) * 32;  // wave N offset in tile
  const int m_block = blockIdx.x * BM;
  const int n_block = blockIdx.y * BN;

  // ---- A staging: row = tid>>1, 16 contiguous K values at col (tid&1)*16 ----
  const int arow = tid >> 1;
  const int acol = (tid & 1) * 16;
  const int am   = m_block + arow;
  const int img  = am / (OH * OW);          // one division, outside loop
  const int pix  = am - img * OH * OW;
  const int oh   = pix / OW;
  const int ow   = pix - oh * OW;
  const int ih0  = oh * stride - 1;
  const int iw0  = ow * stride - 1;
  const size_t imgrow = (size_t)img * IH;

  int aky = 0, akx = 0, ac0 = acol;         // k0 = acol: ky=kx=0, c=acol

  const bf16* wtile = wgt + (size_t)n_block * K;   // B tile origin (row pitch K)
#if !HAVE_TDM
  const int brow = tid >> 2;
  const int bcol = (tid & 3) * 8;
  const bf16* wptr = wtile + (size_t)brow * K + bcol;
  v8bf rb;
#endif

  const int half = lane >> 4;
  const int lr   = lane & 15;

  v8f acc[2][2];
#pragma unroll
  for (int i = 0; i < 2; i++)
#pragma unroll
    for (int j = 0; j < 2; j++)
#pragma unroll
      for (int r = 0; r < 8; r++) acc[i][j][r] = 0.0f;

  v8bf ra0, ra1;

  // ---- prologue: chunk 0 -> LDS buf 0 ----
#if HAVE_TDM
  if (wv == 0) tdm_load_b_tile(wtile, &Bs[0][0], K, COUT);
#else
  rb = *(const v8bf*)wptr;
#endif
  {
    int ih = ih0 + aky, iw = iw0 + akx;
#pragma unroll
    for (int j = 0; j < 8; j++) { ra0[j] = (bf16)0.0f; ra1[j] = (bf16)0.0f; }
    if (ih >= 0 && ih < IH && iw >= 0 && iw < IW) {
      const bf16* p = &in[((imgrow + ih) * IW + iw) * CIN + ac0];
      ra0 = *(const v8bf*)p;
      ra1 = *(const v8bf*)(p + 8);
    }
  }
  *(v8bf*)&As[0][arow * LDK + acol]     = ra0;
  *(v8bf*)&As[0][arow * LDK + acol + 8] = ra1;
#if HAVE_TDM
  if (wv == 0) __builtin_amdgcn_s_wait_tensorcnt(0);
#else
  *(v8bf*)&Bs[0][brow * LDK + bcol] = rb;
#endif
  __syncthreads();

  int buf = 0;
  for (int kc = 0; kc < K; kc += BK) {
    const bool notlast = (kc + BK < K);
    if (notlast) {
#if HAVE_TDM
      // issue DMA for next weight tile early: overlaps with the 4 WMMAs below
      if (wv == 0) tdm_load_b_tile(wtile + (kc + BK), &Bs[buf ^ 1][0], K, COUT);
#endif
      // incremental decode of next chunk's (ky,kx,c); CIN % 32 == 0
      ac0 += BK;
      if (ac0 >= CIN) { ac0 -= CIN; akx++; if (akx == 3) { akx = 0; aky++; } }
      int ih = ih0 + aky, iw = iw0 + akx;
#pragma unroll
      for (int j = 0; j < 8; j++) { ra0[j] = (bf16)0.0f; ra1[j] = (bf16)0.0f; }
      if (ih >= 0 && ih < IH && iw >= 0 && iw < IW) {
        const bf16* p = &in[((imgrow + ih) * IW + iw) * CIN + ac0];
        ra0 = *(const v8bf*)p;
        ra1 = *(const v8bf*)(p + 8);
      }
#if !HAVE_TDM
      wptr += BK;
      rb = *(const v8bf*)wptr;
#endif
    }

    // ---- compute on LDS[buf]: 2 A frags x 2 B frags -> 4 WMMAs ----
    const bf16* Ab = &As[buf][0];
    const bf16* Bb = &Bs[buf][0];

    v16bf afrag[2], bfrag[2];
#pragma unroll
    for (int i = 0; i < 2; i++) {
      v8bf lo = *(const v8bf*)&Ab[(mw + i * 16 + lr) * LDK + half * 8];
      v8bf hi = *(const v8bf*)&Ab[(mw + i * 16 + lr) * LDK + 16 + half * 8];
      afrag[i] = __builtin_shufflevector(lo, hi, 0, 1, 2, 3, 4, 5, 6, 7,
                                         8, 9, 10, 11, 12, 13, 14, 15);
    }
#pragma unroll
    for (int j = 0; j < 2; j++) {
      v8bf lo = *(const v8bf*)&Bb[(nw + j * 16 + lr) * LDK + half * 8];
      v8bf hi = *(const v8bf*)&Bb[(nw + j * 16 + lr) * LDK + 16 + half * 8];
      bfrag[j] = __builtin_shufflevector(lo, hi, 0, 1, 2, 3, 4, 5, 6, 7,
                                         8, 9, 10, 11, 12, 13, 14, 15);
    }
#pragma unroll
    for (int i = 0; i < 2; i++)
#pragma unroll
      for (int j = 0; j < 2; j++)
        acc[i][j] = __builtin_amdgcn_wmma_f32_16x16x32_bf16(
            false, afrag[i], false, bfrag[j], (short)0, acc[i][j], false, false);

    if (notlast) {
      *(v8bf*)&As[buf ^ 1][arow * LDK + acol]     = ra0;
      *(v8bf*)&As[buf ^ 1][arow * LDK + acol + 8] = ra1;
#if HAVE_TDM
      if (wv == 0) __builtin_amdgcn_s_wait_tensorcnt(0);
#else
      *(v8bf*)&Bs[buf ^ 1][brow * LDK + bcol] = rb;
#endif
      __syncthreads();
      buf ^= 1;
    }
  }

  // ---- epilogue: D element (M = r + 8*half, N = lane&15) per sub-tile ----
#pragma unroll
  for (int j = 0; j < 2; j++) {
    const int nglob = n_block + nw + j * 16 + lr;
    const float bv  = bias[nglob];
#pragma unroll
    for (int i = 0; i < 2; i++) {
#pragma unroll
      for (int r = 0; r < 8; r++) {
        int mg = m_block + mw + i * 16 + r + 8 * half;
        float v = acc[i][j][r] + bv;
        if (LRELU) v = (v >= 0.0f) ? v : 0.2f * v;
        size_t o = (size_t)mg * COUT + nglob;
        if (RESID) v += (float)resid[o];   // read-then-write same element: safe
        out[o] = (bf16)v;
      }
    }
  }
}

// ---------------------------------------------------------------------------
// Final conv 256->1 @8x8 (pad=1) + spatial mean per image.
// ---------------------------------------------------------------------------
__global__ __launch_bounds__(64) void final_conv_mean(
    const bf16* __restrict__ in, const bf16* __restrict__ w,
    const float* __restrict__ bias, float* __restrict__ imgmean)
{
  __shared__ float red[64];
  int img = blockIdx.x;
  int tid = threadIdx.x;
  int oh = tid >> 3, ow = tid & 7;
  float sum = bias[0];
  for (int ky = 0; ky < 3; ky++) {
    int ih = oh - 1 + ky;
    if (ih < 0 || ih > 7) continue;
    for (int kx = 0; kx < 3; kx++) {
      int iw = ow - 1 + kx;
      if (iw < 0 || iw > 7) continue;
      const bf16* ip = &in[(((size_t)img * 8 + ih) * 8 + iw) * 256];
      const bf16* wp = &w[(ky * 3 + kx) * 256];
      for (int c = 0; c < 256; c++) sum += (float)wp[c] * (float)ip[c];
    }
  }
  red[tid] = sum;
  __syncthreads();
  for (int s = 32; s > 0; s >>= 1) {
    if (tid < s) red[tid] += red[tid + s];
    __syncthreads();
  }
  if (tid == 0) imgmean[img] = red[0] * (1.0f / 64.0f);
}

__global__ void combine_out(const float* __restrict__ imgmean, float* __restrict__ out)
{
  int b = threadIdx.x;
  if (b < 16)
    out[b] = 0.25f * (imgmean[b * 4] + imgmean[b * 4 + 1] +
                      imgmean[b * 4 + 2] + imgmean[b * 4 + 3]);
}

// ---------------------------------------------------------------------------
// Host orchestration
// ---------------------------------------------------------------------------
struct OpDesc { char type; int cin, cout, stride; };
static const OpDesc g_ops[17] = {
  {'c',9,64,1},{'c',64,64,1},{'r',64,64,1},{'r',64,64,1},{'c',64,128,2},
  {'r',128,128,1},{'r',128,128,1},{'c',128,128,2},{'r',128,128,1},{'r',128,128,1},
  {'c',128,128,2},{'r',128,128,1},{'r',128,128,1},{'c',128,256,2},{'r',256,256,1},
  {'r',256,256,1},{'c',256,1,1}
};

extern "C" void kernel_launch(void* const* d_in, const int* in_sizes, int n_in,
                              void* d_out, int out_size, void* d_ws, size_t ws_size,
                              hipStream_t stream) {
  const float* x       = (const float*)d_in[0];
  const float* input_x = (const float*)d_in[2];
  const int*   bboxes  = (const int*)d_in[4];
  // params flattened after the 5 tensors: (w, b, u) per conv, 27 convs

  char* ws = (char*)d_ws;
  size_t off = 0;
  auto alloc = [&](size_t bytes) -> char* {
    char* p = ws + off;
    off += (bytes + 255) & ~(size_t)255;
    return p;
  };

  bf16*  patches = (bf16*)alloc((size_t)64 * 128 * 128 * 32 * sizeof(bf16));
  bf16*  bufA    = (bf16*)alloc((size_t)64 * 128 * 128 * 64 * sizeof(bf16));
  bf16*  bufB    = (bf16*)alloc((size_t)64 * 128 * 128 * 64 * sizeof(bf16));
  float* imgmean = (float*)alloc(64 * sizeof(float));

  int pcin[27], pcinp[27], pcout[27];
  int np = 0;
  for (int oi = 0; oi < 17; oi++) {
    const OpDesc& od = g_ops[oi];
    if (od.type == 'c') {
      pcin[np] = od.cin; pcinp[np] = (od.cin == 9) ? 32 : od.cin; pcout[np] = od.cout; np++;
    } else {
      pcin[np] = od.cin;  pcinp[np] = od.cin;  pcout[np] = od.cout; np++;
      pcin[np] = od.cout; pcinp[np] = od.cout; pcout[np] = od.cout; np++;
    }
  }
  bf16* wbf[27];
  for (int j = 0; j < 27; j++)
    wbf[j] = (bf16*)alloc((size_t)pcout[j] * 9 * pcinp[j] * sizeof(bf16));

  for (int j = 0; j < 27; j++) {
    const float* w = (const float*)d_in[5 + 3 * j + 0];
    const float* u = (const float*)d_in[5 + 3 * j + 2];
    sn_transpose<<<dim3(1), dim3(256), 0, stream>>>(w, u, wbf[j], pcout[j], pcin[j], pcinp[j]);
  }

  build_patches<<<dim3((64 * 128 * 128) / 256), dim3(256), 0, stream>>>(
      x, input_x, bboxes, patches);

  bf16* cur  = patches;
  int   curC = 32;
  int   Hcur = 128;
  int   pj   = 0;
  auto other = [&](bf16* c) -> bf16* { return (c == bufA) ? bufB : bufA; };

  for (int oi = 0; oi < 16; oi++) {
    const OpDesc& od = g_ops[oi];
    if (od.type == 'c') {
      int OHs = Hcur / od.stride;
      bf16* outb = other(cur);
      const float* bptr = (const float*)d_in[5 + 3 * pj + 1];
      dim3 grid((64 * OHs * OHs) / BM, od.cout / BN);
      conv3x3_wmma<true, false><<<grid, dim3(256), 0, stream>>>(
          cur, wbf[pj], bptr, (const bf16*)nullptr, outb,
          Hcur, Hcur, OHs, OHs, curC, od.cout, od.stride);
      cur = outb; curC = od.cout; Hcur = OHs;
      pj++;
    } else {
      bf16* tmp = other(cur);
      const float* b0 = (const float*)d_in[5 + 3 * pj + 1];
      const float* b1 = (const float*)d_in[5 + 3 * (pj + 1) + 1];
      dim3 grid((64 * Hcur * Hcur) / BM, od.cout / BN);
      // r = lrelu(conv1(h))
      conv3x3_wmma<true, false><<<grid, dim3(256), 0, stream>>>(
          cur, wbf[pj], b0, (const bf16*)nullptr, tmp,
          Hcur, Hcur, Hcur, Hcur, curC, od.cout, 1);
      // h = h + conv2(r)   (in-place over h: element read-once-then-written)
      conv3x3_wmma<false, true><<<grid, dim3(256), 0, stream>>>(
          tmp, wbf[pj + 1], b1, cur, cur,
          Hcur, Hcur, Hcur, Hcur, od.cout, od.cout, 1);
      pj += 2;
    }
  }

  const float* bfin = (const float*)d_in[5 + 3 * pj + 1];
  final_conv_mean<<<dim3(64), dim3(64), 0, stream>>>(cur, wbf[pj], bfin, imgmean);
  combine_out<<<dim3(1), dim3(16), 0, stream>>>(imgmean, (float*)d_out);
}
